// CPRRouter_28312424415702
// MI455X (gfx1250) — compile-verified, matrix operand order checked by reading, not run on hardware
//
#include <hip/hip_runtime.h>

// ---------------------------------------------------------------------------
// CPR router for MI455X (gfx1250, wave32, WMMA).
//   logits = (H/||H||) @ (P/||P||)^T ; softmax over 64 experts; top-8.
// v2: drop the 257KB LDS proto cache (WGP$/L2 serve the 4KB/K-step hot
// window); shrink blocks to 4 waves and launch 256 blocks so multiple
// blocks stack per WGP and hide HBM latency on the hidden-state stream.
// ---------------------------------------------------------------------------

typedef __attribute__((ext_vector_type(16))) __bf16 v16bf;
typedef __attribute__((ext_vector_type(8)))  __bf16 v8bf;
typedef __attribute__((ext_vector_type(8)))  float  v8f;

#define TOKENS 16384
#define HID    2048
#define NEXP   64
#define TOPK   8

#define WAVES_PER_BLOCK 4
#define TOK_PER_WAVE    16
#define TOK_PER_BLOCK   (WAVES_PER_BLOCK * TOK_PER_WAVE)   // 64
#define NBLOCKS         (TOKENS / TOK_PER_BLOCK)           // 256

#define LOGIT_STRIDE 68     // 64 + 4 float pad: conflict-free spill / scan

// ---------------------------------------------------------------------------
// Kernel 0: L2-normalize protos, emit bf16 [64][2048] into workspace.
// ---------------------------------------------------------------------------
__global__ __launch_bounds__(256) void cpr_proto_norm(const float* __restrict__ proto,
                                                      unsigned short* __restrict__ proto_bf) {
    __shared__ float part[8];
    __shared__ float s_inv;
    const int e = blockIdx.x;
    const int t = threadIdx.x;
    const float* row = proto + (size_t)e * HID;

    float v[8];
    float ss = 0.f;
#pragma unroll
    for (int i = 0; i < 8; ++i) { v[i] = row[t + i * 256]; ss += v[i] * v[i]; }
#pragma unroll
    for (int o = 16; o > 0; o >>= 1) ss += __shfl_xor(ss, o, 32);
    if ((t & 31) == 0) part[t >> 5] = ss;
    __syncthreads();
    if (t == 0) {
        float s = 0.f;
#pragma unroll
        for (int w = 0; w < 8; ++w) s += part[w];
        s_inv = 1.f / fmaxf(sqrtf(s), 1e-12f);
    }
    __syncthreads();
    const float inv = s_inv;
    __bf16* out = (__bf16*)proto_bf;
#pragma unroll
    for (int i = 0; i < 8; ++i)
        out[(size_t)e * HID + t + i * 256] = (__bf16)(v[i] * inv);
}

// ---------------------------------------------------------------------------
// Kernel 1: GEMM (bf16 WMMA) + row-norm fold + softmax + top-8.
// Block = 4 waves; wave owns a 16-token M tile and all 4 N tiles (64 experts).
// A fragments: fp32 hidden streamed from HBM, converted to bf16 in-register
// (norms accumulated in fp32). B fragments: bf16 normalized protos from the
// workspace — 4KB hot window per K-step, served by WGP$/L2.
// ---------------------------------------------------------------------------
__global__ __launch_bounds__(128) void cpr_router(const float* __restrict__ hidden,
                                                  const unsigned short* __restrict__ proto_bf,
                                                  float* __restrict__ out) {
    __shared__ float lds_logit[WAVES_PER_BLOCK * TOK_PER_WAVE * LOGIT_STRIDE];

    const int tid  = threadIdx.x;
    const int wave = tid >> 5;
    const int lane = tid & 31;

    const int tok0 = blockIdx.x * TOK_PER_BLOCK + wave * TOK_PER_WAVE;
    const int m    = lane & 15;         // row (A) / column (B) within tile
    const int hi   = lane >> 4;
    const int off  = hi << 3;           // K sub-offset per 16-bit WMMA layout

    const float*  hrow = hidden + (size_t)(tok0 + m) * HID + off;
    const __bf16* pbb  = (const __bf16*)proto_bf + (size_t)m * HID + off;

    v8f acc[4] = {};                    // 4 N-tiles of 16x16 f32
    float ss = 0.f;                     // per-lane partial sum of squares (fp32)

#pragma unroll 2
    for (int kb = 0; kb < HID; kb += 32) {
        // A: 16 fp32 per lane, two contiguous 8-elem chunks (layout-exact)
        float4 f0 = *(const float4*)(hrow + kb);
        float4 f1 = *(const float4*)(hrow + kb + 4);
        float4 f2 = *(const float4*)(hrow + kb + 16);
        float4 f3 = *(const float4*)(hrow + kb + 20);
        __builtin_prefetch(hrow + kb + 256, 0, 0);

        ss += f0.x*f0.x + f0.y*f0.y + f0.z*f0.z + f0.w*f0.w;
        ss += f1.x*f1.x + f1.y*f1.y + f1.z*f1.z + f1.w*f1.w;
        ss += f2.x*f2.x + f2.y*f2.y + f2.z*f2.z + f2.w*f2.w;
        ss += f3.x*f3.x + f3.y*f3.y + f3.z*f3.z + f3.w*f3.w;

        v16bf a;
        a[0]=(__bf16)f0.x;  a[1]=(__bf16)f0.y;  a[2]=(__bf16)f0.z;  a[3]=(__bf16)f0.w;
        a[4]=(__bf16)f1.x;  a[5]=(__bf16)f1.y;  a[6]=(__bf16)f1.z;  a[7]=(__bf16)f1.w;
        a[8]=(__bf16)f2.x;  a[9]=(__bf16)f2.y;  a[10]=(__bf16)f2.z; a[11]=(__bf16)f2.w;
        a[12]=(__bf16)f3.x; a[13]=(__bf16)f3.y; a[14]=(__bf16)f3.z; a[15]=(__bf16)f3.w;

#pragma unroll
        for (int t = 0; t < 4; ++t) {
            const __bf16* bp = pbb + (size_t)(t * 16) * HID + kb;
            v8bf b0 = *(const v8bf*)bp;          // K = kb+off   .. +7
            v8bf b1 = *(const v8bf*)(bp + 16);   // K = kb+off+16.. +23
            v16bf b = __builtin_shufflevector(b0, b1,
                       0,1,2,3,4,5,6,7,8,9,10,11,12,13,14,15);
            acc[t] = __builtin_amdgcn_wmma_f32_16x16x32_bf16(
                         false, a, false, b, (short)0, acc[t], false, false);
        }
    }

    // --- fold in 1/||h|| (exact fp32 norm) ---------------------------------
    ss += __shfl_xor(ss, 16, 32);                     // combine K halves of row m
    const float inv = 1.f / fmaxf(sqrtf(ss), 1e-12f);
    float nrm[8];
#pragma unroll
    for (int j = 0; j < 8; ++j)
        nrm[j] = __shfl(inv, j + off, 32);            // norm of C-row (j + 8*hi)

    // --- spill scaled logits: [16 tok][64 exp] per wave, padded ------------
    float* lrow = lds_logit + wave * TOK_PER_WAVE * LOGIT_STRIDE;
#pragma unroll
    for (int t = 0; t < 4; ++t)
#pragma unroll
        for (int j = 0; j < 8; ++j)
            lrow[(j + 8 * hi) * LOGIT_STRIDE + t * 16 + m] = acc[t][j] * nrm[j];

    // --- softmax + top-8 (lanes 0..15, one token each; wave-local LDS) -----
    if (lane < 16) {
        float* v = lrow + lane * LOGIT_STRIDE;
        float mx = -3.4e38f;
        for (int c = 0; c < NEXP; ++c) mx = fmaxf(mx, v[c]);
        float s = 0.f;
        for (int c = 0; c < NEXP; ++c) {
            float e = __expf(v[c] - mx);
            s += e;
            v[c] = e;
        }
        const float invs = 1.f / s;

        const int tok = tok0 + lane;
        float* outw = out + (size_t)tok * TOPK;
        int*   outi = (int*)out + (size_t)TOKENS * TOPK + (size_t)tok * TOPK;

        unsigned long long sel = 0ull;
        for (int k = 0; k < TOPK; ++k) {
            float bv = -1.f; int bi = 0;
            for (int c = 0; c < NEXP; ++c) {
                if (!((sel >> c) & 1ull)) {
                    float x = v[c];
                    if (x > bv) { bv = x; bi = c; }
                }
            }
            sel |= (1ull << bi);
            outw[k] = bv * invs;
            outi[k] = bi;
        }
    }
}

// ---------------------------------------------------------------------------
extern "C" void kernel_launch(void* const* d_in, const int* in_sizes, int n_in,
                              void* d_out, int out_size, void* d_ws, size_t ws_size,
                              hipStream_t stream) {
    (void)in_sizes; (void)n_in; (void)out_size; (void)ws_size;
    const float* hidden = (const float*)d_in[0];
    const float* proto  = (const float*)d_in[1];
    unsigned short* proto_bf = (unsigned short*)d_ws;   // 64*2048 bf16 = 256 KB

    cpr_proto_norm<<<NEXP, 256, 0, stream>>>(proto, proto_bf);
    cpr_router<<<NBLOCKS, WAVES_PER_BLOCK * 32, 0, stream>>>(hidden, proto_bf,
                                                             (float*)d_out);
}
